// RelAttBartAttention_65472481460443
// MI455X (gfx1250) — matrix operand order repaired and checked by reading, not compiled, not run on hardware
//
#include <hip/hip_runtime.h>
#include <hip/hip_bf16.h>

// ---------------------------------------------------------------------------
// RelAtt BART attention for gfx1250 (MI455X).
//   - bf16 WMMA (v_wmma_f32_16x16x32_bf16) for the 4 projection GEMMs
//   - bandwidth-bound rel_k/rel_v streaming with non-temporal loads +
//     global_prefetch, fused score/softmax/weighted-V kernel
// ---------------------------------------------------------------------------

#define EMBED 1024
#define HEADS 16
#define HD 64
#define BATCH 2
#define SEQ 256
#define BH (BATCH * HEADS)
#define MROWS (BATCH * SEQ)   // 512

typedef __attribute__((ext_vector_type(16))) __bf16 v16bf;
typedef __attribute__((ext_vector_type(8)))  float  v8f;

__device__ __forceinline__ __bf16 f2bf(float f) {
    unsigned u = __builtin_bit_cast(unsigned, f);
    unsigned r = u + 0x7FFFu + ((u >> 16) & 1u);   // round to nearest even
    unsigned short h = (unsigned short)(r >> 16);
    return __builtin_bit_cast(__bf16, h);
}

// ---------------------------------------------------------------------------
// float -> bf16 conversion (grid-stride)
// ---------------------------------------------------------------------------
__global__ void cvt_f32_bf16_kernel(const float* __restrict__ src,
                                    __bf16* __restrict__ dst, int n) {
    for (int i = blockIdx.x * blockDim.x + threadIdx.x; i < n;
         i += gridDim.x * blockDim.x)
        dst[i] = f2bf(src[i]);
}

// ---------------------------------------------------------------------------
// C[m,n] = alpha * (sum_k A[m,k] * W[n,k] + bias[n])
// A: M x K bf16 row-major, W: N x K bf16 row-major (we compute A @ W^T).
// One wave computes a 16x16 tile via v_wmma_f32_16x16x32_bf16, K step = 32.
// ---------------------------------------------------------------------------
__global__ void gemm_bf16_wmma_kernel(const __bf16* __restrict__ A,
                                      const __bf16* __restrict__ W,
                                      const float* __restrict__ bias,
                                      float* __restrict__ C,
                                      int Mdim, int Ndim, int Kdim,
                                      float alpha) {
    const int wave = threadIdx.x >> 5;
    const int lane = threadIdx.x & 31;
    const int half = lane >> 4;       // 0: low K-chunk lanes, 1: high
    const int lrow = lane & 15;

    const int ntiles_n = Ndim >> 4;
    const int tile = blockIdx.x * (blockDim.x >> 5) + wave;
    const int total = (Mdim >> 4) * ntiles_n;
    if (tile >= total) return;

    const int tm = (tile / ntiles_n) << 4;   // tile row base (M)
    const int tn = (tile % ntiles_n) << 4;   // tile col base (N)

    v8f acc = {};
    for (int k0 = 0; k0 < Kdim; k0 += 32) {
        // A fragment: 16x32 bf16.  lane<16: M=lrow, K = k0+{0..7,16..23}
        //                          lane>=16: M=lrow, K = k0+{8..15,24..31}
        v16bf af;
        const __bf16* ap = A + (size_t)(tm + lrow) * Kdim + k0 + half * 8;
        #pragma unroll
        for (int j = 0; j < 8; ++j) {
            af[j]     = ap[j];
            af[8 + j] = ap[16 + j];
        }
        // B fragment: 32x16 bf16 (B = W^T, column n == row n of W, contiguous K)
        // lane<16: N=lrow, K=k0+{0..7,16..23} ; lane>=16: N=lrow, K=k0+{8..15,24..31}
        v16bf bf;
        const __bf16* bp = W + (size_t)(tn + lrow) * Kdim + k0 + half * 8;
        #pragma unroll
        for (int j = 0; j < 8; ++j) {
            bf[j]     = bp[j];
            bf[8 + j] = bp[16 + j];
        }

        acc = __builtin_amdgcn_wmma_f32_16x16x32_bf16(
            false, af, false, bf, (short)0, acc, false, false);
    }

    // C/D layout: VGPR i -> (lanes 0-15: M=i, lanes 16-31: M=i+8), N = lane&15
    const int col = tn + lrow;
    const float b = bias[col];
    #pragma unroll
    for (int i = 0; i < 8; ++i) {
        const int row = tm + i + half * 8;
        C[(size_t)row * Ndim + col] = alpha * (acc[i] + b);
    }
}

// ---------------------------------------------------------------------------
// Fused relative attention for one (bh, t):
//   scores[s] = q . (k[s] + rel_k[s,t])   -> softmax over s
//   out[h]    = sum_s p[s] * (v[s][h] + rel_v[s,t][h])
// Block = 256 threads (8 waves). Wave w owns s = w*32..w*32+31; lane owns
// 2 consecutive h values -> every wave-level load is a contiguous 256B line.
// rel_* loads are non-temporal (single-use, 1.07 GB stream).
// ---------------------------------------------------------------------------
__global__ void rel_attn_kernel(const float* __restrict__ Q,   // (B,T,E), prescaled
                                const float* __restrict__ K,   // (B,T,E)
                                const float* __restrict__ V,   // (B,T,E)
                                const float* __restrict__ relK,// (BH,S,T,HD)
                                const float* __restrict__ relV,// (BH,S,T,HD)
                                float* __restrict__ attn)      // (B,T,E)
{
    const int t  = blockIdx.x;
    const int bh = blockIdx.y;
    const int b  = bh / HEADS;
    const int head = bh % HEADS;

    const int tid  = threadIdx.x;
    const int wave = tid >> 5;
    const int lane = tid & 31;

    __shared__ float qs[HD];
    __shared__ float sc[SEQ];
    __shared__ float red[8];
    __shared__ float red2[8];
    __shared__ float partial[8 * HD];

    const size_t qkvBase = ((size_t)(b * SEQ)) * EMBED + head * HD;
    if (tid < HD) qs[tid] = Q[qkvBase + (size_t)t * EMBED + tid];
    __syncthreads();

    const float qx = qs[2 * lane];
    const float qy = qs[2 * lane + 1];

    // ---------------- phase 1: scores ----------------
    const size_t relStride = (size_t)SEQ * HD;                 // per-s stride
    const size_t relBase   = ((size_t)bh * SEQ) * relStride + (size_t)t * HD;

    for (int i = 0; i < 32; ++i) {
        const int s = wave * 32 + i;
        const float* kp  = K + qkvBase + (size_t)s * EMBED + 2 * lane;
        const float* rkp = relK + relBase + (size_t)s * relStride + 2 * lane;
        // prefetch the matching rel_v line for phase 2
        __builtin_prefetch(relV + relBase + (size_t)s * relStride + 2 * lane, 0, 0);

        const float kx  = kp[0], ky = kp[1];
        const float rkx = __builtin_nontemporal_load(rkp);
        const float rky = __builtin_nontemporal_load(rkp + 1);
        float d = qx * (kx + rkx) + qy * (ky + rky);
        #pragma unroll
        for (int m = 16; m > 0; m >>= 1) d += __shfl_xor(d, m, 32);
        if (lane == 0) sc[s] = d;
    }
    __syncthreads();

    // ---------------- softmax over 256 scores ----------------
    float val = sc[tid];
    float mx = val;
    #pragma unroll
    for (int m = 16; m > 0; m >>= 1) mx = fmaxf(mx, __shfl_xor(mx, m, 32));
    if (lane == 0) red[wave] = mx;
    __syncthreads();
    if (tid == 0) {
        float m0 = red[0];
        #pragma unroll
        for (int i = 1; i < 8; ++i) m0 = fmaxf(m0, red[i]);
        red[0] = m0;
    }
    __syncthreads();
    const float e = __expf(val - red[0]);
    float sum = e;
    #pragma unroll
    for (int m = 16; m > 0; m >>= 1) sum += __shfl_xor(sum, m, 32);
    if (lane == 0) red2[wave] = sum;
    __syncthreads();
    if (tid == 0) {
        float s0 = 0.f;
        #pragma unroll
        for (int i = 0; i < 8; ++i) s0 += red2[i];
        red2[0] = s0;
    }
    __syncthreads();
    sc[tid] = e / red2[0];
    __syncthreads();

    // ---------------- phase 2: weighted values ----------------
    float ax = 0.f, ay = 0.f;
    for (int i = 0; i < 32; ++i) {
        const int s = wave * 32 + i;
        const float p = sc[s];
        const float* vp  = V + qkvBase + (size_t)s * EMBED + 2 * lane;
        const float* rvp = relV + relBase + (size_t)s * relStride + 2 * lane;
        const float vx  = vp[0], vy = vp[1];
        const float rvx = __builtin_nontemporal_load(rvp);
        const float rvy = __builtin_nontemporal_load(rvp + 1);
        ax += p * (vx + rvx);
        ay += p * (vy + rvy);
    }
    partial[wave * HD + 2 * lane]     = ax;
    partial[wave * HD + 2 * lane + 1] = ay;
    __syncthreads();

    if (tid < HD) {
        float o = 0.f;
        #pragma unroll
        for (int w = 0; w < 8; ++w) o += partial[w * HD + tid];
        attn[qkvBase + (size_t)t * EMBED + tid] = o;
    }
}

// ---------------------------------------------------------------------------
extern "C" void kernel_launch(void* const* d_in, const int* in_sizes, int n_in,
                              void* d_out, int out_size, void* d_ws, size_t ws_size,
                              hipStream_t stream) {
    (void)in_sizes; (void)n_in; (void)out_size; (void)ws_size;

    const float* hs   = (const float*)d_in[0];
    const float* relK = (const float*)d_in[1];
    const float* relV = (const float*)d_in[2];
    const float* Wq   = (const float*)d_in[3];
    const float* bq   = (const float*)d_in[4];
    const float* Wk   = (const float*)d_in[5];
    const float* bk   = (const float*)d_in[6];
    const float* Wv   = (const float*)d_in[7];
    const float* bv   = (const float*)d_in[8];
    const float* Wo   = (const float*)d_in[9];
    const float* bo   = (const float*)d_in[10];
    float* out = (float*)d_out;

    // workspace carve-up
    char* ws = (char*)d_ws;
    __bf16* hbf  = (__bf16*)ws;  ws += (size_t)MROWS * EMBED * sizeof(__bf16);
    __bf16* wqb  = (__bf16*)ws;  ws += (size_t)EMBED * EMBED * sizeof(__bf16);
    __bf16* wkb  = (__bf16*)ws;  ws += (size_t)EMBED * EMBED * sizeof(__bf16);
    __bf16* wvb  = (__bf16*)ws;  ws += (size_t)EMBED * EMBED * sizeof(__bf16);
    __bf16* wob  = (__bf16*)ws;  ws += (size_t)EMBED * EMBED * sizeof(__bf16);
    float*  Qf   = (float*)ws;   ws += (size_t)MROWS * EMBED * sizeof(float);
    float*  Kf   = (float*)ws;   ws += (size_t)MROWS * EMBED * sizeof(float);
    float*  Vf   = (float*)ws;   ws += (size_t)MROWS * EMBED * sizeof(float);
    float*  attn = (float*)ws;   ws += (size_t)MROWS * EMBED * sizeof(float);
    __bf16* atb  = (__bf16*)ws;  ws += (size_t)MROWS * EMBED * sizeof(__bf16);

    const int nHid = MROWS * EMBED;   // 524288
    const int nW   = EMBED * EMBED;   // 1048576

    cvt_f32_bf16_kernel<<<512, 256, 0, stream>>>(hs, hbf, nHid);
    cvt_f32_bf16_kernel<<<1024, 256, 0, stream>>>(Wq, wqb, nW);
    cvt_f32_bf16_kernel<<<1024, 256, 0, stream>>>(Wk, wkb, nW);
    cvt_f32_bf16_kernel<<<1024, 256, 0, stream>>>(Wv, wvb, nW);
    cvt_f32_bf16_kernel<<<1024, 256, 0, stream>>>(Wo, wob, nW);

    // projections: (512x1024) @ (1024x1024)^T -> 2048 tiles / 8 waves = 256 blocks
    const float scaling = 0.125f;     // HD^-0.5
    gemm_bf16_wmma_kernel<<<256, 256, 0, stream>>>(hbf, wqb, bq, Qf,
                                                   MROWS, EMBED, EMBED, scaling);
    gemm_bf16_wmma_kernel<<<256, 256, 0, stream>>>(hbf, wkb, bk, Kf,
                                                   MROWS, EMBED, EMBED, 1.0f);
    gemm_bf16_wmma_kernel<<<256, 256, 0, stream>>>(hbf, wvb, bv, Vf,
                                                   MROWS, EMBED, EMBED, 1.0f);

    // fused relative attention, one block per (t, bh)
    dim3 agrid(SEQ, BH);
    rel_attn_kernel<<<agrid, 256, 0, stream>>>(Qf, Kf, Vf, relK, relV, attn);

    // output projection
    cvt_f32_bf16_kernel<<<512, 256, 0, stream>>>(attn, atb, nHid);
    gemm_bf16_wmma_kernel<<<256, 256, 0, stream>>>(atb, wob, bo, out,
                                                   MROWS, EMBED, EMBED, 1.0f);
}
